// Preprocess_Line2Window_64072322122005
// MI455X (gfx1250) — matrix operand order repaired and checked by reading, not compiled, not run on hardware
//
#include <hip/hip_runtime.h>
#include <hip/hip_fp16.h>

#define BD 2
#define CD 128
#define HD 60
#define WD 80
#define ND 3072
#define TSCALE 16.0f
#define EPSV 1e-6f

typedef __attribute__((ext_vector_type(16))) _Float16 v16h;
typedef __attribute__((ext_vector_type(8)))  float    v8f;

// ---------------------------------------------------------------------------
// Kernel 1: per-pixel channel l2norm of feature map, scaled by T.
// Threads = pixels; per-c iteration is coalesced across threads (x-contiguous).
// ---------------------------------------------------------------------------
__global__ void k_chan_norm(const float* __restrict__ xf, float* __restrict__ xft) {
  int p = blockIdx.x * blockDim.x + threadIdx.x;
  const int total = BD * HD * WD;
  if (p >= total) return;
  int b = p / (HD * WD);
  int pix = p % (HD * WD);
  const float* src = xf + (size_t)b * CD * HD * WD + pix;
  float ss = 0.f;
#pragma unroll 4
  for (int c = 0; c < CD; ++c) {
    float v = src[(size_t)c * HD * WD];
    ss += v * v;
  }
  float s = TSCALE * rsqrtf(ss + 1e-12f);
  float* dst = xft + (size_t)b * CD * HD * WD + pix;
#pragma unroll 4
  for (int c = 0; c < CD; ++c) dst[(size_t)c * HD * WD] = src[(size_t)c * HD * WD] * s;
}

// ---------------------------------------------------------------------------
// Kernel 2: bilinear sample (align_corners, border clamp) + per-point l2norm.
// One block (128 threads = one channel each) per point. Emits f32 + f16 copies,
// plus denormed coords and valid flag.
// ---------------------------------------------------------------------------
__global__ void k_sample_norm(const float* __restrict__ xf, const float* __restrict__ cn,
                              float* __restrict__ f, _Float16* __restrict__ fh,
                              float* __restrict__ out_coord, float* __restrict__ out_valid,
                              const int* __restrict__ hp, const int* __restrict__ wp) {
  int i = blockIdx.x;            // b*ND + n
  int b = i / ND;
  int c = threadIdx.x;
  float cx = cn[(size_t)i * 2 + 0];
  float cy = cn[(size_t)i * 2 + 1];
  float x = (cx + 1.f) * 0.5f * (float)(WD - 1);
  float y = (cy + 1.f) * 0.5f * (float)(HD - 1);
  float x0f = floorf(x), y0f = floorf(y);
  float wx = x - x0f, wy = y - y0f;
  int x0 = min(max((int)x0f, 0), WD - 1);
  int x1 = min(x0 + 1, WD - 1);
  int y0 = min(max((int)y0f, 0), HD - 1);
  int y1 = min(y0 + 1, HD - 1);
  const float* pl = xf + ((size_t)b * CD + c) * HD * WD;
  float v = pl[y0 * WD + x0] * (1.f - wx) * (1.f - wy) +
            pl[y0 * WD + x1] * wx * (1.f - wy) +
            pl[y1 * WD + x0] * (1.f - wx) * wy +
            pl[y1 * WD + x1] * wx * wy;
  __shared__ float red[128];
  red[c] = v * v;
  __syncthreads();
  for (int s = 64; s > 0; s >>= 1) {
    if (c < s) red[c] += red[c + s];
    __syncthreads();
  }
  float scale = rsqrtf(red[0] + 1e-12f);
  float fv = v * scale;
  f[(size_t)i * CD + c] = fv;
  fh[(size_t)i * CD + c] = (_Float16)fv;
  if (c == 0) out_coord[(size_t)i * 2 + 0] = (cx + 1.f) * 0.5f * (float)(wp[0] - 1);
  if (c == 1) out_coord[(size_t)i * 2 + 1] = (cy + 1.f) * 0.5f * (float)(hp[0] - 1);
  if (c == 2) out_valid[i] = 1.0f;
}

// ---------------------------------------------------------------------------
// Kernel 3: S[b][n][m] = T * sum_c f1[n][c]*f2[m][c] via v_wmma_f32_16x16x32_f16.
// One wave = one 16x16 tile, 4 WMMAs over K=128. Fragment layouts per ISA 7.12.2.
// ---------------------------------------------------------------------------
__global__ void k_gemm_wmma(const _Float16* __restrict__ f1h, const _Float16* __restrict__ f2h,
                            float* __restrict__ S) {
  int lane = threadIdx.x & 31;
  int wv = threadIdx.x >> 5;
  int b = blockIdx.z;
  int M0 = blockIdx.y * 16;
  int N0 = (blockIdx.x * 4 + wv) * 16;
  int mn = lane & 15;            // A-row (M) and B-col (N) index for this lane
  int hi = lane >> 4;
  const _Float16* arow = f1h + ((size_t)b * ND + M0 + mn) * CD;
  const _Float16* brow = f2h + ((size_t)b * ND + N0 + mn) * CD;
  v8f acc = {};
#pragma unroll
  for (int k0 = 0; k0 < CD; k0 += 32) {
    v16h a, bb;
    // A 16x32: lane<16 -> K {0..7,16..23}; lane>=16 -> K {8..15,24..31}
    reinterpret_cast<float4*>(&a)[0] =
        *reinterpret_cast<const float4*>(arow + k0 + hi * 8);
    reinterpret_cast<float4*>(&a)[1] =
        *reinterpret_cast<const float4*>(arow + k0 + 16 + hi * 8);
    // B 32x16: lane holds N=lane&15; lane<16 -> K {0..7,16..23}; lane>=16 -> K {8..15,24..31}
    reinterpret_cast<float4*>(&bb)[0] = *reinterpret_cast<const float4*>(brow + k0 + hi * 8);
    reinterpret_cast<float4*>(&bb)[1] = *reinterpret_cast<const float4*>(brow + k0 + 16 + hi * 8);
    acc = __builtin_amdgcn_wmma_f32_16x16x32_f16(false, a, false, bb, (short)0, acc,
                                                 false, false);
  }
  // D layout: VGPR r -> M = M0 + 8*hi + r, N = N0 + (lane&15)
  float* srow = S + (size_t)b * ND * ND + (size_t)(M0 + 8 * hi) * ND + N0 + mn;
#pragma unroll
  for (int r = 0; r < 8; ++r) srow[(size_t)r * ND] = TSCALE * acc[r];
}

// ---------------------------------------------------------------------------
// Kernel 4: row softmax (axis=m) + coordinate moments -> g1, g1n, std1.
// One 256-thread block per row; S rows are L2-resident.
// ---------------------------------------------------------------------------
__global__ void k_row_softmax(const float* __restrict__ S, const float* __restrict__ c2n,
                              float* __restrict__ g1, float* __restrict__ g1n,
                              float* __restrict__ std1, const int* __restrict__ h2p,
                              const int* __restrict__ w2p) {
  int b = blockIdx.y, n = blockIdx.x, tid = threadIdx.x;
  const float* row = S + ((size_t)b * ND + n) * ND;
  __shared__ float red[256];
  float mx = -3.402823466e38f;
  for (int m = tid; m < ND; m += 256) mx = fmaxf(mx, row[m]);
  red[tid] = mx;
  __syncthreads();
  for (int s = 128; s > 0; s >>= 1) {
    if (tid < s) red[tid] = fmaxf(red[tid], red[tid + s]);
    __syncthreads();
  }
  mx = red[0];
  __syncthreads();
  float se = 0.f, sx = 0.f, sy = 0.f, sxx = 0.f, syy = 0.f;
  for (int m = tid; m < ND; m += 256) {
    float e = __expf(row[m] - mx);
    float cx = c2n[((size_t)b * ND + m) * 2 + 0];
    float cy = c2n[((size_t)b * ND + m) * 2 + 1];
    se += e; sx += e * cx; sy += e * cy; sxx += e * cx * cx; syy += e * cy * cy;
  }
  float vals[5] = {se, sx, sy, sxx, syy};
  float res[5];
  for (int k = 0; k < 5; ++k) {
    red[tid] = vals[k];
    __syncthreads();
    for (int s = 128; s > 0; s >>= 1) {
      if (tid < s) red[tid] += red[tid + s];
      __syncthreads();
    }
    res[k] = red[0];
    __syncthreads();
  }
  if (tid == 0) {
    float inv = 1.f / res[0];
    float gx = res[1] * inv, gy = res[2] * inv;
    float exx = res[3] * inv, eyy = res[4] * inv;
    int h2 = h2p[0], w2 = w2p[0];
    size_t o = (size_t)b * ND + n;
    g1n[o * 2 + 0] = gx;
    g1n[o * 2 + 1] = gy;
    g1[o * 2 + 0] = (gx + 1.f) * 0.5f * (float)(w2 - 1);
    g1[o * 2 + 1] = (gy + 1.f) * 0.5f * (float)(h2 - 1);
    std1[o] = sqrtf(fmaxf(exx - gx * gx, EPSV)) + sqrtf(fmaxf(eyy - gy * gy, EPSV));
  }
}

// ---------------------------------------------------------------------------
// Kernel 5: column softmax (axis=n). One thread per column; reads are coalesced
// across threads (contiguous m within each row of S).
// ---------------------------------------------------------------------------
__global__ void k_col_softmax(const float* __restrict__ S, const float* __restrict__ c1n,
                              float* __restrict__ g2, float* __restrict__ g2n,
                              float* __restrict__ std2, const int* __restrict__ h1p,
                              const int* __restrict__ w1p) {
  int m = blockIdx.x * blockDim.x + threadIdx.x;
  int b = blockIdx.y;
  if (m >= ND) return;
  const float* col = S + (size_t)b * ND * ND + m;
  float mx = -3.402823466e38f;
  for (int n = 0; n < ND; ++n) mx = fmaxf(mx, col[(size_t)n * ND]);
  float se = 0.f, sx = 0.f, sy = 0.f, sxx = 0.f, syy = 0.f;
  for (int n = 0; n < ND; ++n) {
    float e = __expf(col[(size_t)n * ND] - mx);
    float cx = c1n[((size_t)b * ND + n) * 2 + 0];
    float cy = c1n[((size_t)b * ND + n) * 2 + 1];
    se += e; sx += e * cx; sy += e * cy; sxx += e * cx * cx; syy += e * cy * cy;
  }
  float inv = 1.f / se;
  float gx = sx * inv, gy = sy * inv;
  int h1 = h1p[0], w1 = w1p[0];
  size_t o = (size_t)b * ND + m;
  g2n[o * 2 + 0] = gx;
  g2n[o * 2 + 1] = gy;
  g2[o * 2 + 0] = (gx + 1.f) * 0.5f * (float)(w1 - 1);
  g2[o * 2 + 1] = (gy + 1.f) * 0.5f * (float)(h1 - 1);
  std2[o] = sqrtf(fmaxf(sxx * inv - gx * gx, EPSV)) + sqrtf(fmaxf(syy * inv - gy * gy, EPSV));
}

// ---------------------------------------------------------------------------
// Kernel 6: 9x9 window expected-correspondence search.
// One block per point; f1 vector staged in LDS; one thread per window sample.
// ---------------------------------------------------------------------------
__global__ void k_window(const float* __restrict__ f, const float* __restrict__ fmapt,
                         const float* __restrict__ center, float* __restrict__ wout,
                         float* __restrict__ wstd, const int* __restrict__ hp,
                         const int* __restrict__ wp) {
  int i = blockIdx.x;            // b*ND + n
  int b = i / ND;
  int tid = threadIdx.x;
  __shared__ float f1s[CD];
  __shared__ float logit[81];
  f1s[tid] = f[(size_t)i * CD + tid];
  __syncthreads();
  float cx = center[(size_t)i * 2 + 0];
  float cy = center[(size_t)i * 2 + 1];
  if (tid < 81) {
    float ox = -0.125f + (float)(tid % 9) * 0.03125f;
    float oy = -0.125f + (float)(tid / 9) * 0.03125f;
    float wcx = cx + ox, wcy = cy + oy;
    float x = (wcx + 1.f) * 0.5f * (float)(WD - 1);
    float y = (wcy + 1.f) * 0.5f * (float)(HD - 1);
    float x0f = floorf(x), y0f = floorf(y);
    float wx = x - x0f, wy = y - y0f;
    int x0 = min(max((int)x0f, 0), WD - 1);
    int x1 = min(x0 + 1, WD - 1);
    int y0 = min(max((int)y0f, 0), HD - 1);
    int y1 = min(y0 + 1, HD - 1);
    float w00 = (1.f - wx) * (1.f - wy), w01 = wx * (1.f - wy);
    float w10 = (1.f - wx) * wy, w11 = wx * wy;
    const float* base = fmapt + (size_t)b * CD * HD * WD;
    int i00 = y0 * WD + x0, i01 = y0 * WD + x1, i10 = y1 * WD + x0, i11 = y1 * WD + x1;
    float dot = 0.f;
#pragma unroll 4
    for (int c = 0; c < CD; ++c) {
      const float* p = base + (size_t)c * HD * WD;
      dot += f1s[c] * (p[i00] * w00 + p[i01] * w01 + p[i10] * w10 + p[i11] * w11);
    }
    logit[tid] = dot;
  }
  __syncthreads();
  if (tid == 0) {
    float mx = -3.402823466e38f;
    for (int s = 0; s < 81; ++s) mx = fmaxf(mx, logit[s]);
    float se = 0.f, sx = 0.f, sy = 0.f, sxx = 0.f, syy = 0.f;
    for (int s = 0; s < 81; ++s) {
      float e = __expf(logit[s] - mx);
      float ox = -0.125f + (float)(s % 9) * 0.03125f;
      float oy = -0.125f + (float)(s / 9) * 0.03125f;
      float wcx = cx + ox, wcy = cy + oy;
      se += e; sx += e * wcx; sy += e * wcy; sxx += e * wcx * wcx; syy += e * wcy * wcy;
    }
    float inv = 1.f / se;
    float ex = sx * inv, ey = sy * inv;
    int h = hp[0], w = wp[0];
    wout[(size_t)i * 2 + 0] = (ex + 1.f) * 0.5f * (float)(w - 1);
    wout[(size_t)i * 2 + 1] = (ey + 1.f) * 0.5f * (float)(h - 1);
    wstd[i] = sqrtf(fmaxf(sxx * inv - ex * ex, EPSV)) + sqrtf(fmaxf(syy * inv - ey * ey, EPSV));
  }
}

// ---------------------------------------------------------------------------
extern "C" void kernel_launch(void* const* d_in, const int* in_sizes, int n_in,
                              void* d_out, int out_size, void* d_ws, size_t ws_size,
                              hipStream_t stream) {
  const float* xf1 = (const float*)d_in[0];
  const float* xf2 = (const float*)d_in[1];
  const float* c1n = (const float*)d_in[2];
  const float* c2n = (const float*)d_in[3];
  const int* h1 = (const int*)d_in[4];
  const int* w1 = (const int*)d_in[5];
  const int* h2 = (const int*)d_in[6];
  const int* w2 = (const int*)d_in[7];

  float* out = (float*)d_out;
  const int PN2 = BD * ND * 2;   // 12288
  const int PN1 = BD * ND;       // 6144
  float* o_coord1 = out;
  float* o_coord2 = o_coord1 + PN2;
  float* o_g1     = o_coord2 + PN2;
  float* o_g2     = o_g1 + PN2;
  float* o_w1     = o_g2 + PN2;
  float* o_w2     = o_w1 + PN2;
  float* o_g1n    = o_w2 + PN2;
  float* o_g2n    = o_g1n + PN2;
  float* o_std1   = o_g2n + PN2;
  float* o_std2   = o_std1 + PN1;
  float* o_wstd1  = o_std2 + PN1;
  float* o_wstd2  = o_wstd1 + PN1;
  float* o_valid1 = o_wstd2 + PN1;
  float* o_valid2 = o_valid1 + PN1;

  // workspace carve (256B aligned regions)
  char* wsp = (char*)d_ws;
  auto carve = [&](size_t bytes) -> void* {
    void* p = (void*)wsp;
    wsp += (bytes + 255) & ~(size_t)255;
    return p;
  };
  float*    xf1t = (float*)carve((size_t)BD * CD * HD * WD * 4);
  float*    xf2t = (float*)carve((size_t)BD * CD * HD * WD * 4);
  float*    f1   = (float*)carve((size_t)BD * ND * CD * 4);
  float*    f2   = (float*)carve((size_t)BD * ND * CD * 4);
  _Float16* f1h  = (_Float16*)carve((size_t)BD * ND * CD * 2);
  _Float16* f2h  = (_Float16*)carve((size_t)BD * ND * CD * 2);
  float*    S    = (float*)carve((size_t)BD * ND * ND * 4);

  // 1) channel-normalized (xT) feature maps
  {
    int total = BD * HD * WD;
    int blocks = (total + 255) / 256;
    k_chan_norm<<<blocks, 256, 0, stream>>>(xf1, xf1t);
    k_chan_norm<<<blocks, 256, 0, stream>>>(xf2, xf2t);
  }
  // 2) point features f1/f2 (+ coord1/coord2 + valid flags)
  k_sample_norm<<<BD * ND, 128, 0, stream>>>(xf1, c1n, f1, f1h, o_coord1, o_valid1, h1, w1);
  k_sample_norm<<<BD * ND, 128, 0, stream>>>(xf2, c2n, f2, f2h, o_coord2, o_valid2, h2, w2);
  // 3) S = T * f1 f2^T via WMMA: grid (192/4 tile-cols, 192 tile-rows, B)
  {
    dim3 g(ND / 16 / 4, ND / 16, BD);
    k_gemm_wmma<<<g, 128, 0, stream>>>(f1h, f2h, S);
  }
  // 4) row softmax -> g1, g1n, std1
  {
    dim3 g(ND, BD);
    k_row_softmax<<<g, 256, 0, stream>>>(S, c2n, o_g1, o_g1n, o_std1, h2, w2);
  }
  // 5) column softmax -> g2, g2n, std2
  {
    dim3 g((ND + 255) / 256, BD);
    k_col_softmax<<<g, 256, 0, stream>>>(S, c1n, o_g2, o_g2n, o_std2, h1, w1);
  }
  // 6) window searches: side1 uses f1 against xf2t at g1n; side2 mirrored.
  k_window<<<BD * ND, 128, 0, stream>>>(f1, xf2t, o_g1n, o_w1, o_wstd1, h2, w2);
  k_window<<<BD * ND, 128, 0, stream>>>(f2, xf1t, o_g2n, o_w2, o_wstd2, h1, w1);
}
